// EmformerDistillModel_86560770884143
// MI455X (gfx1250) — matrix operand order repaired and checked by reading, not compiled, not run on hardware
//
#include <hip/hip_runtime.h>
#include <hip/hip_bf16.h>
#include <cstdint>

// ---------------- model constants ----------------
#define B_       2
#define T_U_     1024
#define D_       80
#define H_       8
#define FFN_     2048
#define SEG_     32
#define RC_      8
#define LC_      50
#define NLAYERS_ 4
#define NSEG_    32
#define RCT_     256          // NSEG*RC
#define L_       1280         // RCT + T_U
#define DH_      10
#define OUT_DIM_ 768
#define KP_      96           // K=80 padded to multiple of 32
#define SCALE_   0.31622776601683794f  // DH^-0.5

typedef __bf16   bf16x16 __attribute__((ext_vector_type(16)));
typedef float    f32x8   __attribute__((ext_vector_type(8)));
typedef uint32_t u32x4   __attribute__((ext_vector_type(4)));

union BfFrag { bf16x16 v; u32x4 q[2]; uint32_t u[8]; };

__device__ __forceinline__ unsigned short f2bf(float f) {
  union { float f; uint32_t u; } c; c.f = f;
  uint32_t u = c.u;
  return (unsigned short)((u + 0x7FFFu + ((u >> 16) & 1u)) >> 16);
}

// ---------------- gather: x = concat(mel[RC_IDX], mel[:T_U]) ----------------
__global__ void gather_kernel(const float* __restrict__ mel, float* __restrict__ x) {
  int idx = blockIdx.x * blockDim.x + threadIdx.x;
  const int total = B_ * L_ * D_;
  if (idx >= total) return;
  int d = idx % D_;
  int l = (idx / D_) % L_;
  int b = idx / (L_ * D_);
  int src;
  if (l < RCT_) {
    int seg = l / RC_, pos = l % RC_;
    src = (seg < NSEG_ - 1) ? (seg + 1) * SEG_ + pos : T_U_ + pos;
  } else {
    src = l - RCT_;
  }
  x[idx] = mel[((size_t)b * (T_U_ + RC_) + src) * D_ + d];
}

// ---------------- layernorm (one wave per row); bf16-padded and/or f32 out ----
__global__ void ln_kernel(const float* __restrict__ in, const float* __restrict__ w,
                          const float* __restrict__ b, unsigned short* __restrict__ outbf,
                          float* __restrict__ outf, int rows) {
  int wid  = (blockIdx.x * blockDim.x + threadIdx.x) >> 5;
  int lane = threadIdx.x & 31;
  if (wid >= rows) return;
  const float* row = in + (size_t)wid * D_;
  float s = 0.f, s2 = 0.f;
  for (int d = lane; d < D_; d += 32) { float v = row[d]; s += v; s2 += v * v; }
  for (int o = 16; o; o >>= 1) { s += __shfl_xor(s, o); s2 += __shfl_xor(s2, o); }
  float mean = s * (1.0f / D_);
  float var  = s2 * (1.0f / D_) - mean * mean;
  float rstd = rsqrtf(var + 1e-5f);
  for (int d = lane; d < KP_; d += 32) {
    float v = 0.f;
    if (d < D_) v = (row[d] - mean) * rstd * w[d] + b[d];
    if (outbf) outbf[(size_t)wid * KP_ + d] = f2bf(v);
    if (outf && d < D_) outf[(size_t)wid * D_ + d] = v;
  }
}

// ---------------- fp32 -> bf16 (K padded to KP_) --------------------------------
__global__ void conv_act_kernel(const float* __restrict__ in, unsigned short* __restrict__ out,
                                int rows, int K) {
  int idx = blockIdx.x * blockDim.x + threadIdx.x;
  int total = rows * KP_;
  if (idx >= total) return;
  int k = idx % KP_, r = idx / KP_;
  out[idx] = (k < K) ? f2bf(in[(size_t)r * K + k]) : (unsigned short)0;
}

// ---------------- weight convert+transpose: W[K][N] f32 -> Wt[N][Kp] bf16 ------
__global__ void conv_wt_kernel(const float* __restrict__ W, unsigned short* __restrict__ Wt,
                               int K, int N, int Kp) {
  int idx = blockIdx.x * blockDim.x + threadIdx.x;
  int total = N * Kp;
  if (idx >= total) return;
  int k = idx % Kp, n = idx / Kp;
  Wt[idx] = (k < K) ? f2bf(W[(size_t)k * N + n]) : (unsigned short)0;
}

// ---------------- generic bf16 WMMA GEMM: C = (A@W + bias)*alpha + resid -------
// A: [rows][Kp] bf16 (row index mapped via Mb/Lb/row_off), Wt: [N][Kp] bf16.
__global__ void gemm_kernel(const unsigned short* __restrict__ A,
                            const unsigned short* __restrict__ Wt,
                            const float* __restrict__ bias,
                            const float* __restrict__ resid,
                            float* __restrict__ C,
                            int Mtiles, int Ntiles, int Kp, int N,
                            int Mb, int Lb, int row_off, float alpha) {
  int wid  = (blockIdx.x * (blockDim.x >> 5)) + (threadIdx.x >> 5);
  int lane = threadIdx.x & 31;
  if (wid >= Mtiles * Ntiles) return;
  int tm = wid / Ntiles, tn = wid % Ntiles;
  int half = lane >> 4, l15 = lane & 15;

  int r    = tm * 16 + l15;
  int grow = (r / Mb) * Lb + row_off + (r % Mb);
  const uint32_t* arp = (const uint32_t*)(A + (size_t)grow * Kp);
  int ncol = tn * 16 + l15;
  const uint32_t* brp = (const uint32_t*)(Wt + (size_t)ncol * Kp);

  f32x8 acc = {0.f, 0.f, 0.f, 0.f, 0.f, 0.f, 0.f, 0.f};
  for (int k0 = 0; k0 < Kp; k0 += 32) {
    BfFrag a, bm;
    const uint32_t* pa = arp + ((k0 + half * 8) >> 1);   // A: 16-bit 16x32 layout
    a.q[0] = *(const u32x4*)pa;
    a.q[1] = *(const u32x4*)(pa + 8);
    const uint32_t* pb = brp + ((k0 + half * 8) >> 1);   // B: 16-bit 32x16 layout (same k order as A)
    bm.q[0] = *(const u32x4*)pb;
    bm.q[1] = *(const u32x4*)(pb + 8);
    if (k0 + 32 < Kp) __builtin_prefetch(pa + 16, 0, 1);
    acc = __builtin_amdgcn_wmma_f32_16x16x32_bf16(false, a.v, false, bm.v,
                                                  (short)0, acc, false, false);
  }
  float bv = bias ? bias[ncol] : 0.f;
  for (int v = 0; v < 8; v++) {
    int m = tm * 16 + v + 8 * half;
    float outv = (acc[v] + bv) * alpha;
    if (resid) {
      int gm = (m / Mb) * Lb + row_off + (m % Mb);
      outv += resid[(size_t)gm * N + ncol];
    }
    C[(size_t)m * N + ncol] = outv;
  }
}

// ---------------- masked block attention (one WG per (b,seg), wave per head) ---
__global__ void attn_kernel(const float* __restrict__ q, const float* __restrict__ kv,
                            float* __restrict__ att) {
  int lane = threadIdx.x & 31;
  int h    = threadIdx.x >> 5;                // 0..7
  int b    = blockIdx.x / NSEG_;
  int i    = blockIdx.x % NSEG_;
  int seg_start = i * SEG_ - LC_; if (seg_start < 0) seg_start = 0;
  int nu = (i + 1) * SEG_ - seg_start;        // utterance window width
  int nk = RC_ + nu;                          // total keys (<=90)

  float kr[3][DH_], vr[3][DH_];
  bool  val[3];
  for (int c = 0; c < 3; c++) {
    int j = lane + 32 * c;
    val[c] = j < nk;
    int gr = 0;
    if (val[c]) gr = (j < RC_) ? (i * RC_ + j) : (RCT_ + seg_start + (j - RC_));
    const float* kp = kv + ((size_t)(b * L_ + gr)) * (2 * D_) + h * DH_;
    const float* vp = kp + D_;
    for (int d = 0; d < DH_; d++) {
      kr[c][d] = val[c] ? kp[d] : 0.f;
      vr[c][d] = val[c] ? vp[d] : 0.f;
    }
  }
  for (int qi = 0; qi < RC_ + SEG_; qi++) {
    int qrow = (qi < RC_) ? (i * RC_ + qi) : (RCT_ + i * SEG_ + (qi - RC_));
    const float* qp = q + ((size_t)(b * L_ + qrow)) * D_ + h * DH_;
    float qv[DH_];
    for (int d = 0; d < DH_; d++) qv[d] = qp[d];
    float s[3];
    for (int c = 0; c < 3; c++) {
      float acc = 0.f;
      for (int d = 0; d < DH_; d++) acc += qv[d] * kr[c][d];
      s[c] = val[c] ? acc : -1e30f;
    }
    float m = fmaxf(s[0], fmaxf(s[1], s[2]));
    for (int o = 16; o; o >>= 1) m = fmaxf(m, __shfl_xor(m, o));
    float p[3], sum = 0.f;
    for (int c = 0; c < 3; c++) { p[c] = val[c] ? __expf(s[c] - m) : 0.f; sum += p[c]; }
    for (int o = 16; o; o >>= 1) sum += __shfl_xor(sum, o);
    float inv = 1.f / sum;
    float mine = 0.f;
    for (int d = 0; d < DH_; d++) {
      float a = p[0] * vr[0][d] + p[1] * vr[1][d] + p[2] * vr[2][d];
      for (int o = 16; o; o >>= 1) a += __shfl_xor(a, o);
      if (lane == d) mine = a * inv;
    }
    if (lane < DH_) att[((size_t)(b * L_ + qrow)) * D_ + h * DH_ + lane] = mine;
  }
}

// ---------------- fused FFN: out = resid + relu(A@W1+b1)@W2 + b2 ---------------
// A: [rows][KP_] bf16 ; W1t: [FFN][KP_] ; W2t: [D][FFN] ; one wave per 16 rows.
// The 16x96 bf16 A-tile is staged into LDS with CDNA5 async-to-LDS DMA
// (GLOBAL_LOAD_ASYNC_TO_LDS_B128, tracked by ASYNCcnt), then fragments are
// re-read 3x from LDS instead of 3x from global.
__global__ void ffn_kernel(const unsigned short* __restrict__ A,
                           const unsigned short* __restrict__ W1t,
                           const float* __restrict__ b1,
                           const unsigned short* __restrict__ W2t,
                           const float* __restrict__ b2,
                           const float* __restrict__ resid,
                           float* __restrict__ out) {
  __shared__ __align__(16) unsigned short lds[16 * 32];    // relu tile re-layout
  __shared__ __align__(16) unsigned short aTile[16 * KP_]; // staged A rows (3 KB)
  int lane = threadIdx.x & 31;
  int half = lane >> 4, l15 = lane & 15;
  int tm = blockIdx.x;

  // ---- async stage A tile: 16 rows * 192 B = 192 x b128 chunks, 6 per lane ----
  {
    const char* gbase = (const char*)(A + (size_t)tm * 16 * KP_);
    uint32_t ldsBase = (uint32_t)(uintptr_t)(&aTile[0]);
    for (int i = 0; i < 6; i++) {
      int t = lane + 32 * i;                       // 16-byte chunk id, 0..191
      uint32_t ldsOff = ldsBase + (uint32_t)t * 16u;
      unsigned long long g = (unsigned long long)(uintptr_t)(gbase + (size_t)t * 16);
      asm volatile("global_load_async_to_lds_b128 %0, %1, off"
                   :: "v"(ldsOff), "v"(g) : "memory");
    }
    asm volatile("s_wait_asynccnt 0x0" ::: "memory");
  }
  __syncthreads();

  // resident A fragments for this 16-row block (K = 96 -> 3 frags), from LDS
  BfFrag af[3];
  const uint32_t* arp = (const uint32_t*)(aTile + (size_t)l15 * KP_);
  for (int t = 0; t < 3; t++) {
    const uint32_t* pa = arp + ((t * 32 + half * 8) >> 1);
    af[t].q[0] = *(const u32x4*)pa;
    af[t].q[1] = *(const u32x4*)(pa + 8);
  }
  f32x8 acc[5];
  for (int j = 0; j < 5; j++) acc[j] = (f32x8){0.f,0.f,0.f,0.f,0.f,0.f,0.f,0.f};

  for (int f0 = 0; f0 < FFN_; f0 += 32) {
    f32x8 t0 = {0.f,0.f,0.f,0.f,0.f,0.f,0.f,0.f};
    f32x8 t1 = {0.f,0.f,0.f,0.f,0.f,0.f,0.f,0.f};
    int n0 = f0 + l15, n1 = f0 + 16 + l15;
    const uint32_t* b1p0 = (const uint32_t*)(W1t + (size_t)n0 * KP_);
    const uint32_t* b1p1 = (const uint32_t*)(W1t + (size_t)n1 * KP_);
    for (int t = 0; t < 3; t++) {
      BfFrag bf0, bf1;
      const uint32_t* p0 = b1p0 + ((t * 32 + half * 8) >> 1);
      bf0.q[0] = *(const u32x4*)p0;  bf0.q[1] = *(const u32x4*)(p0 + 8);
      const uint32_t* p1 = b1p1 + ((t * 32 + half * 8) >> 1);
      bf1.q[0] = *(const u32x4*)p1;  bf1.q[1] = *(const u32x4*)(p1 + 8);
      t0 = __builtin_amdgcn_wmma_f32_16x16x32_bf16(false, af[t].v, false, bf0.v,
                                                   (short)0, t0, false, false);
      t1 = __builtin_amdgcn_wmma_f32_16x16x32_bf16(false, af[t].v, false, bf1.v,
                                                   (short)0, t1, false, false);
    }
    float bias0 = b1[n0], bias1 = b1[n1];
    for (int v = 0; v < 8; v++) {               // C layout -> LDS row-major bf16
      int m = v + 8 * half;
      lds[m * 32 + l15]      = f2bf(fmaxf(t0[v] + bias0, 0.f));
      lds[m * 32 + 16 + l15] = f2bf(fmaxf(t1[v] + bias1, 0.f));
    }
    __syncthreads();
    BfFrag a2;                                   // re-gather as A fragment (16x32)
    const uint32_t* lp = (const uint32_t*)(lds + l15 * 32 + half * 8);
    a2.q[0] = *(const u32x4*)lp;
    a2.q[1] = *(const u32x4*)(lp + 8);
    __syncthreads();
    for (int j = 0; j < 5; j++) {
      BfFrag b2f;
      const uint32_t* p2 = (const uint32_t*)(W2t + (size_t)(j * 16 + l15) * FFN_
                                             + f0 + half * 8);
      b2f.q[0] = *(const u32x4*)p2;  b2f.q[1] = *(const u32x4*)(p2 + 8);
      acc[j] = __builtin_amdgcn_wmma_f32_16x16x32_bf16(false, a2.v, false, b2f.v,
                                                       (short)0, acc[j], false, false);
    }
  }
  for (int j = 0; j < 5; j++) {
    int n = j * 16 + l15;
    float bb = b2[n];
    for (int v = 0; v < 8; v++) {
      int m = tm * 16 + v + 8 * half;
      out[(size_t)m * D_ + n] = acc[j][v] + bb + resid[(size_t)m * D_ + n];
    }
  }
}

// ---------------- lengths passthrough into tuple tail --------------------------
__global__ void tail_kernel(const int* __restrict__ lens, int* __restrict__ out) {
  int t = threadIdx.x;
  if (t < B_) out[t] = lens[t];
}

// =================================================================================
extern "C" void kernel_launch(void* const* d_in, const int* in_sizes, int n_in,
                              void* d_out, int out_size, void* d_ws, size_t ws_size,
                              hipStream_t stream) {
  (void)in_sizes; (void)n_in; (void)out_size; (void)ws_size;

  const float* mel      = (const float*)d_in[0];
  const int*   lens     = (const int*)  d_in[1];
  const float* ln_in_w  = (const float*)d_in[2];
  const float* ln_in_b  = (const float*)d_in[3];
  const float* wq       = (const float*)d_in[4];
  const float* bq       = (const float*)d_in[5];
  const float* wkv      = (const float*)d_in[6];
  const float* bkv      = (const float*)d_in[7];
  const float* wo       = (const float*)d_in[8];
  const float* bo       = (const float*)d_in[9];
  const float* ln_ff_w  = (const float*)d_in[10];
  const float* ln_ff_b  = (const float*)d_in[11];
  const float* w1       = (const float*)d_in[12];
  const float* b1       = (const float*)d_in[13];
  const float* w2       = (const float*)d_in[14];
  const float* b2       = (const float*)d_in[15];
  const float* ln_out_w = (const float*)d_in[16];
  const float* ln_out_b = (const float*)d_in[17];
  const float* proj_w   = (const float*)d_in[18];
  const float* proj_b   = (const float*)d_in[19];

  char* ws = (char*)d_ws;
  float*          x     = (float*)(ws + 0);        // [B,L,D]
  float*          res   = (float*)(ws + 819200);   // [B,L,D]
  float*          y     = (float*)(ws + 1638400);  // [B,L,D]
  float*          qbuf  = (float*)(ws + 2457600);  // [B,L,D]
  float*          att   = (float*)(ws + 3276800);  // [B,L,D]
  float*          kvb   = (float*)(ws + 4096000);  // [B,L,2D]
  unsigned short* xnbf  = (unsigned short*)(ws + 5734400); // [B*L,96] bf16
  unsigned short* attbf = (unsigned short*)(ws + 6225920); // [B*L,96] bf16
  unsigned short* wqT   = (unsigned short*)(ws + 6717440); // [80][96]
  unsigned short* wkvT  = (unsigned short*)(ws + 6732800); // [160][96]
  unsigned short* woT   = (unsigned short*)(ws + 6763520); // [80][96]
  unsigned short* w1T   = (unsigned short*)(ws + 6778880); // [2048][96]
  unsigned short* w2T   = (unsigned short*)(ws + 7172096); // [80][2048]
  unsigned short* projT = (unsigned short*)(ws + 7499776); // [768][96]

  const int ROWS   = B_ * L_;      // 2560
  const int MT     = ROWS / 16;    // 160
  const int LN_GRID = (ROWS + 7) / 8;   // 8 waves per 256-thread block

  { int tot = ROWS * D_;
    gather_kernel<<<(tot + 255) / 256, 256, 0, stream>>>(mel, x); }

  for (int layer = 0; layer < NLAYERS_; ++layer) {
    const float* lwq  = wq  + (size_t)layer * D_ * D_;
    const float* lwkv = wkv + (size_t)layer * D_ * 2 * D_;
    const float* lwo  = wo  + (size_t)layer * D_ * D_;
    const float* lw1  = w1  + (size_t)layer * D_ * FFN_;
    const float* lw2  = w2  + (size_t)layer * FFN_ * D_;
    const float* lbq  = bq  + (size_t)layer * D_;
    const float* lbkv = bkv + (size_t)layer * 2 * D_;
    const float* lbo  = bo  + (size_t)layer * D_;
    const float* lb1  = b1  + (size_t)layer * FFN_;
    const float* lb2  = b2  + (size_t)layer * D_;
    const float* liw  = ln_in_w  + (size_t)layer * D_;
    const float* lib  = ln_in_b  + (size_t)layer * D_;
    const float* lfw  = ln_ff_w  + (size_t)layer * D_;
    const float* lfb  = ln_ff_b  + (size_t)layer * D_;
    const float* low  = ln_out_w + (size_t)layer * D_;
    const float* lob  = ln_out_b + (size_t)layer * D_;

    // weight conversion (bf16, transposed, K padded)
    conv_wt_kernel<<<(D_   * KP_  + 255) / 256, 256, 0, stream>>>(lwq,  wqT,  D_,   D_,   KP_);
    conv_wt_kernel<<<(2*D_ * KP_  + 255) / 256, 256, 0, stream>>>(lwkv, wkvT, D_,   2*D_, KP_);
    conv_wt_kernel<<<(D_   * KP_  + 255) / 256, 256, 0, stream>>>(lwo,  woT,  D_,   D_,   KP_);
    conv_wt_kernel<<<(FFN_ * KP_  + 255) / 256, 256, 0, stream>>>(lw1,  w1T,  D_,   FFN_, KP_);
    conv_wt_kernel<<<(D_   * FFN_ + 255) / 256, 256, 0, stream>>>(lw2,  w2T,  FFN_, D_,   FFN_);

    // xn = LN(x) -> bf16
    ln_kernel<<<LN_GRID, 256, 0, stream>>>(x, liw, lib, xnbf, nullptr, ROWS);
    // q = (xn@wq + bq)*scale ; kv = xn@wkv + bkv
    gemm_kernel<<<(MT * 5  + 3) / 4, 128, 0, stream>>>(xnbf, wqT,  lbq,  nullptr, qbuf,
                                                       MT, 5,  KP_, D_,    L_, L_, 0, SCALE_);
    gemm_kernel<<<(MT * 10 + 3) / 4, 128, 0, stream>>>(xnbf, wkvT, lbkv, nullptr, kvb,
                                                       MT, 10, KP_, 2*D_,  L_, L_, 0, 1.0f);
    // block attention
    attn_kernel<<<B_ * NSEG_, 256, 0, stream>>>(qbuf, kvb, att);
    // res = att@wo + bo + x
    conv_act_kernel<<<(ROWS * KP_ + 255) / 256, 256, 0, stream>>>(att, attbf, ROWS, D_);
    gemm_kernel<<<(MT * 5 + 3) / 4, 128, 0, stream>>>(attbf, woT, lbo, x, res,
                                                      MT, 5, KP_, D_, L_, L_, 0, 1.0f);
    // y = res + relu(LN(res)@w1+b1)@w2 + b2   (fused FFN)
    ln_kernel<<<LN_GRID, 256, 0, stream>>>(res, lfw, lfb, xnbf, nullptr, ROWS);
    ffn_kernel<<<MT, 32, 0, stream>>>(xnbf, w1T, lb1, w2T, lb2, res, y);
    // x = LN(y)  (next-layer input / final u)
    ln_kernel<<<LN_GRID, 256, 0, stream>>>(y, low, lob, nullptr, x, ROWS);
  }

  // final projection: out = u @ proj_w + proj_b, u = x[:, RCT:]
  conv_wt_kernel<<<(OUT_DIM_ * KP_ + 255) / 256, 256, 0, stream>>>(proj_w, projT, D_, OUT_DIM_, KP_);
  conv_act_kernel<<<(ROWS * KP_ + 255) / 256, 256, 0, stream>>>(x, xnbf, ROWS, D_);
  {
    const int MTp = (B_ * T_U_) / 16;   // 128 row tiles
    const int NTp = OUT_DIM_ / 16;      // 48 col tiles
    gemm_kernel<<<(MTp * NTp + 3) / 4, 128, 0, stream>>>(xnbf, projT, proj_b, nullptr,
                                                         (float*)d_out,
                                                         MTp, NTp, KP_, OUT_DIM_,
                                                         T_U_, L_, RCT_, 1.0f);
  }
  tail_kernel<<<1, 32, 0, stream>>>(lens, (int*)d_out + (size_t)B_ * T_U_ * OUT_DIM_);
}
